// Model_80023830659638
// MI455X (gfx1250) — matrix operand order repaired and checked by reading, not compiled, not run on hardware
//
#include <hip/hip_runtime.h>
#include <hip/hip_bf16.h>
#include <math.h>

// ---------------------------------------------------------------------------
// MI455X (gfx1250, wave32) implementation.
//
// Pipeline (all intermediates in d_ws, L2-resident):
//   1. stft_kernel        : windowed table-DFT magnitude -> h0 (B,1024,128)
//   2. enc_layer_kernel x8: dilated residual channel-mix GEMMs via
//                           v_wmma_f32_16x16x32_f16, async double-buffered
//                           LDS tiles (global_load_async_to_lds_b32)
//   3. latent_kernel      : time-mean -> 16-dim latent, unit-norm
//   4. proj_wmma_kernel   : 256x1024 up-projection via WMMA -> enc (B,256,128)
//   5. topk_kernel        : flattened top-16 (channel, frame) per batch
//   6. dry_kernel         : events = 16 scaled shifted atoms (impulse conv
//                           collapses analytically; no FFT needed)
//   7. mlp_kernel         : room-softmax + mix-sigmoid tiny MLPs
//   8. impulse_kernel     : impulse = room @ impulses
//   9. wet_mix_kernel     : causal direct convolution dry (*) impulse,
//                           LDS-tiled (async-staged), fused dry/wet mix
// ---------------------------------------------------------------------------

typedef __attribute__((ext_vector_type(16))) _Float16 v16h;
typedef __attribute__((ext_vector_type(8)))  _Float16 v8h;
typedef __attribute__((ext_vector_type(8)))  float    v8f;

#define NSAMP 32768
#define NFRM  128
#define NCH   1024
#define HSTR  36   // padded LDS row stride in f32: 144B (16B-aligned, conflict-free)

#if __has_builtin(__builtin_amdgcn_global_load_async_to_lds_b32) && \
    __has_builtin(__builtin_amdgcn_s_wait_asynccnt)
#define USE_ASYNC 1
#else
#define USE_ASYNC 0
#endif

typedef __attribute__((address_space(1))) int g_int;
typedef __attribute__((address_space(3))) int l_int;

// Stage a 32(K) x 128(T) f32 tile into LDS, transposed to [t][k].
// Each of 256 threads moves 16 elements -> 16 async ops per wave.
__device__ __forceinline__ void stage_tile(const float* __restrict__ src,
                                           float* dst, int k0, int tid)
{
#pragma unroll
    for (int e = 0; e < 16; ++e) {
        int idx = tid + e * 256;
        int kk = idx >> 7, t = idx & 127;
#if USE_ASYNC
        __builtin_amdgcn_global_load_async_to_lds_b32(
            (g_int*)(src + (k0 + kk) * NFRM + t),
            (l_int*)(dst + t * HSTR + kk),
            0, 0);
#else
        dst[t * HSTR + kk] = src[(k0 + kk) * NFRM + t];
#endif
    }
}

// Wait for all but (optionally) the most recent 16 async ops of this wave.
__device__ __forceinline__ void wait_async_tile(int keep_newest_batch)
{
#if USE_ASYNC
    if (keep_newest_batch) __builtin_amdgcn_s_wait_asynccnt(16);
    else                   __builtin_amdgcn_s_wait_asynccnt(0);
#else
    (void)keep_newest_batch;
#endif
}

// Build a 16-element f16 fragment from 16 consecutive f32 in LDS
// (contiguous 64B -> ds_load_b128 x4 + cvt).
__device__ __forceinline__ v16h frag_from_lds(const float* rowp)
{
    v16h r;
#pragma unroll
    for (int e = 0; e < 16; ++e) r[e] = (_Float16)rowp[e];
    return r;
}

// ---------------------------------------------------------------------------
// 1. STFT magnitude: one block per (frame, batch). Hann window + table DFT.
// ---------------------------------------------------------------------------
__global__ void __launch_bounds__(256) stft_kernel(const float* __restrict__ x,
                                                   float* __restrict__ h0)
{
    __shared__ float fr[2048];
    __shared__ float twc[2048];
    __shared__ float tws[2048];
    const int f = blockIdx.x;      // frame 0..127
    const int b = blockIdx.y;      // batch
    const int tid = threadIdx.x;

    for (int i = tid; i < 2048; i += 256) {
        float ang = 6.28318530717958647f * (float)i * (1.0f / 2048.0f);
        twc[i] = __cosf(ang);
        tws[i] = __sinf(ang);
        int g = f * 256 + i;
        float v = (g < NSAMP) ? x[(size_t)b * NSAMP + g] : 0.0f;
        float wv = 0.5f - 0.5f * __cosf(6.28318530717958647f * (float)i / 2047.0f);
        fr[i] = v * wv;
    }
    __syncthreads();

    for (int kk = 0; kk < 4; ++kk) {
        int k = tid + kk * 256;            // bin 0..1023
        float re = 0.0f, im = 0.0f;
        int j = 0;
        for (int i = 0; i < 2048; ++i) {
            float v = fr[i];
            re = fmaf(v, twc[j], re);
            im = fmaf(v, tws[j], im);
            j = (j + k) & 2047;
        }
        h0[((size_t)b * NCH + k) * NFRM + f] = sqrtf(re * re + im * im);
    }
}

// ---------------------------------------------------------------------------
// 2. One dilated residual encoder layer:
//    hout[o,t] = hin[o,t] + leaky( sum_i w0[o,i]h[i,t] + w1[o,i]h[i,t+d] + b[o] )
//    M=1024, N=128, K=1024 (x2 taps) per batch, f16 WMMA, f32 accumulate.
//    Async double-buffered activation tiles overlap DMA with WMMA.
// ---------------------------------------------------------------------------
__global__ void __launch_bounds__(256) enc_layer_kernel(
    const float* __restrict__ hin, const float* __restrict__ w2tap,
    const float* __restrict__ bias, float* __restrict__ hout, int d)
{
    __shared__ float hs[2][128][HSTR];     // 36 KB double buffer, [t][k] f32

    const int b    = blockIdx.y;
    const int m0   = blockIdx.x * 128;
    const int lane = threadIdx.x & 31;
    const int wave = threadIdx.x >> 5;
    const int mrow = m0 + wave * 16;
    const int lhalf = lane >> 4;                 // 0 or 1 (lane half)
    const float* __restrict__ hb = hin + (size_t)b * NCH * NFRM;
    const float2* __restrict__ wp = (const float2*)w2tap; // {tap0,tap1} pairs

    v8f acc[8];
#pragma unroll
    for (int j = 0; j < 8; ++j) acc[j] = (v8f)0.0f;

    const int am = mrow + (lane & 15);           // A-matrix row (out channel)

    stage_tile(hb, &hs[0][0][0], 0, threadIdx.x);   // prologue: tile 0
    int ibuf = 0;
    for (int k0 = 0; k0 < NCH; k0 += 32, ibuf ^= 1) {
        const int has_next = (k0 + 32 < NCH);
        if (has_next)                                // issue next tile's DMA
            stage_tile(hb, &hs[ibuf ^ 1][0][0], k0 + 32, threadIdx.x);
        wait_async_tile(has_next);                   // current tile landed
        __syncthreads();
        const float (*cur)[HSTR] = hs[ibuf];

        // A fragments (lanes0-15 K=0..15, lanes16-31 K=16..31)
        v16h a0, a1;
        const int kA = k0 + lhalf * 16;
#pragma unroll
        for (int e = 0; e < 8; ++e) {
            float2 p = wp[(size_t)am * NCH + (kA + e)];
            float2 q = wp[(size_t)am * NCH + (kA + 8 + e)];
            a0[e]     = (_Float16)p.x;  a1[e]     = (_Float16)p.y;
            a0[8 + e] = (_Float16)q.x;  a1[8 + e] = (_Float16)q.y;
        }

        // B fragments (lane%16 = column, lane-half selects K 0..15 / 16..31)
        const int kB = lhalf * 16;
#pragma unroll
        for (int j = 0; j < 8; ++j) {
            const int n = j * 16 + (lane & 15);      // time column
            v16h b0 = frag_from_lds(&cur[n][kB]);
            v16h b1 = (v16h)(_Float16)0.0f;          // anti-causal tap, zero pad
            if (n + d < NFRM) b1 = frag_from_lds(&cur[n + d][kB]);
            acc[j] = __builtin_amdgcn_wmma_f32_16x16x32_f16(
                false, a0, false, b0, (short)0, acc[j], false, false);
            acc[j] = __builtin_amdgcn_wmma_f32_16x16x32_f16(
                false, a1, false, b1, (short)0, acc[j], false, false);
        }
        __syncthreads();                             // tile fully consumed
    }

    // epilogue: bias + leaky_relu(0.2) + residual
#pragma unroll
    for (int j = 0; j < 8; ++j) {
        const int n = j * 16 + (lane & 15);
#pragma unroll
        for (int r = 0; r < 8; ++r) {
            const int m = mrow + (lhalf << 3) + r;   // C/D layout: VGPR r -> M
            float v = acc[j][r] + bias[m];
            v = v > 0.0f ? v : 0.2f * v;
            hout[((size_t)b * NCH + m) * NFRM + n] = hb[(size_t)m * NFRM + n] + v;
        }
    }
}

// ---------------------------------------------------------------------------
// 3. Time-mean -> latent(16) -> unit norm. One block per batch.
// ---------------------------------------------------------------------------
__global__ void __launch_bounds__(256) latent_kernel(
    const float* __restrict__ h, const float* __restrict__ lat_w,
    const float* __restrict__ lat_b, float* __restrict__ dense)
{
    __shared__ float zs[1024];
    __shared__ float dl[16];
    const int b = blockIdx.x, tid = threadIdx.x;
    for (int c = tid; c < NCH; c += 256) {
        const float* row = h + ((size_t)b * NCH + c) * NFRM;
        float s = 0.0f;
#pragma unroll 4
        for (int t = 0; t < NFRM; ++t) s += row[t];
        zs[c] = s * (1.0f / 128.0f);
    }
    __syncthreads();
    if (tid < 16) {
        float s = lat_b[tid];
        for (int c = 0; c < NCH; ++c) s += zs[c] * lat_w[c * 16 + tid];
        dl[tid] = s;
    }
    __syncthreads();
    if (tid == 0) {
        float n2 = 0.0f;
        for (int j = 0; j < 16; ++j) n2 += dl[j] * dl[j];
        float inv = 1.0f / (sqrtf(n2) + 1e-8f);
        for (int j = 0; j < 16; ++j) dense[b * 16 + j] = dl[j] * inv;
    }
}

// ---------------------------------------------------------------------------
// 4. 1x1 up-projection: enc[o,t] = sum_c up_w[o,c] h[c,t] + up_b[o]
//    M=256, N=128, K=1024 per batch — same async WMMA scheme.
// ---------------------------------------------------------------------------
__global__ void __launch_bounds__(256) proj_wmma_kernel(
    const float* __restrict__ hin, const float* __restrict__ upw,
    const float* __restrict__ upb, float* __restrict__ out)
{
    __shared__ float hs[2][128][HSTR];
    const int b    = blockIdx.y;
    const int m0   = blockIdx.x * 128;           // 0 or 128
    const int lane = threadIdx.x & 31;
    const int wave = threadIdx.x >> 5;
    const int mrow = m0 + wave * 16;
    const int lhalf = lane >> 4;
    const float* __restrict__ hb = hin + (size_t)b * NCH * NFRM;

    v8f acc[8];
#pragma unroll
    for (int j = 0; j < 8; ++j) acc[j] = (v8f)0.0f;

    const int am = mrow + (lane & 15);

    stage_tile(hb, &hs[0][0][0], 0, threadIdx.x);
    int ibuf = 0;
    for (int k0 = 0; k0 < NCH; k0 += 32, ibuf ^= 1) {
        const int has_next = (k0 + 32 < NCH);
        if (has_next)
            stage_tile(hb, &hs[ibuf ^ 1][0][0], k0 + 32, threadIdx.x);
        wait_async_tile(has_next);
        __syncthreads();
        const float (*cur)[HSTR] = hs[ibuf];

        v16h a0;
        const int kA = k0 + lhalf * 16;
#pragma unroll
        for (int e = 0; e < 8; ++e) {
            a0[e]     = (_Float16)upw[(size_t)am * NCH + kA + e];
            a0[8 + e] = (_Float16)upw[(size_t)am * NCH + kA + 8 + e];
        }
        const int kB = lhalf * 16;
#pragma unroll
        for (int j = 0; j < 8; ++j) {
            const int n = j * 16 + (lane & 15);
            v16h b0 = frag_from_lds(&cur[n][kB]);
            acc[j] = __builtin_amdgcn_wmma_f32_16x16x32_f16(
                false, a0, false, b0, (short)0, acc[j], false, false);
        }
        __syncthreads();
    }
#pragma unroll
    for (int j = 0; j < 8; ++j) {
        const int n = j * 16 + (lane & 15);
#pragma unroll
        for (int r = 0; r < 8; ++r) {
            const int m = mrow + (lhalf << 3) + r;
            out[((size_t)b * 256 + m) * NFRM + n] = acc[j][r] + upb[m];
        }
    }
}

// ---------------------------------------------------------------------------
// 5. top-16 over flattened (256 ch x 128 frames). One block per batch.
//    flat index in the reference's "full" array maps: ch = e>>7, t = (e&127)*256
// ---------------------------------------------------------------------------
__global__ void __launch_bounds__(256) topk_kernel(
    const float* __restrict__ enc, float* __restrict__ vals,
    int* __restrict__ chs, int* __restrict__ ts)
{
    __shared__ float bv[256];
    __shared__ int   bi[256];
    __shared__ int   chosen[16];
    __shared__ float chval[16];
    const int b = blockIdx.x, tid = threadIdx.x;
    const float* e = enc + (size_t)b * 256 * NFRM;

    for (int it = 0; it < 16; ++it) {
        float best = -3.4e38f; int besti = 0;
        for (int idx = tid; idx < 256 * NFRM; idx += 256) {
            bool used = false;
            for (int q = 0; q < it; ++q) used |= (chosen[q] == idx);
            float v = used ? -3.4e38f : e[idx];
            if (v > best) { best = v; besti = idx; }
        }
        bv[tid] = best; bi[tid] = besti;
        __syncthreads();
        for (int s = 128; s > 0; s >>= 1) {
            if (tid < s && bv[tid + s] > bv[tid]) {
                bv[tid] = bv[tid + s]; bi[tid] = bi[tid + s];
            }
            __syncthreads();
        }
        if (tid == 0) { chosen[it] = bi[0]; chval[it] = bv[0]; }
        __syncthreads();
    }
    if (tid < 16) {
        int idx = chosen[tid];
        vals[b * 16 + tid] = chval[tid];
        chs[b * 16 + tid]  = idx >> 7;          // channel -> atom index
        ts[b * 16 + tid]   = (idx & 127) << 8;  // frame * STEP
    }
}

// ---------------------------------------------------------------------------
// 6. dry[n] = sum_k vals[k] * atoms[ch_k][n - t_k]
// ---------------------------------------------------------------------------
__global__ void __launch_bounds__(256) dry_kernel(
    const float* __restrict__ atoms, const float* __restrict__ vals,
    const int* __restrict__ chs, const int* __restrict__ ts,
    float* __restrict__ dry)
{
    const int b = blockIdx.y;
    const int n = blockIdx.x * 256 + threadIdx.x;
    float s = 0.0f;
#pragma unroll
    for (int k = 0; k < 16; ++k) {
        int t = ts[b * 16 + k];
        if (n >= t)
            s += vals[b * 16 + k] *
                 atoms[(size_t)chs[b * 16 + k] * NSAMP + (n - t)];
    }
    dry[(size_t)b * NSAMP + n] = s;
}

// ---------------------------------------------------------------------------
// 7. room/mix MLPs (LinearOutputStack 16-wide, LayerNorm, leaky 0.2).
// ---------------------------------------------------------------------------
__global__ void __launch_bounds__(64) mlp_kernel(
    const float* __restrict__ dense,
    const float* __restrict__ rw, const float* __restrict__ rb,
    const float* __restrict__ rg, const float* __restrict__ rbe,
    const float* __restrict__ row_, const float* __restrict__ rob,
    const float* __restrict__ mw, const float* __restrict__ mb,
    const float* __restrict__ mg, const float* __restrict__ mbe,
    const float* __restrict__ mow, const float* __restrict__ mob,
    float* __restrict__ room, float* __restrict__ mix)
{
    __shared__ float z[4][16];
    __shared__ float t2[4][16];
    const int tid = threadIdx.x;
    const int b = tid >> 4, j = tid & 15;

    for (int which = 0; which < 2; ++which) {       // 0 = room, 1 = mix
        const float* W  = which ? mw  : rw;
        const float* Bv = which ? mb  : rb;
        const float* G  = which ? mg  : rg;
        const float* BE = which ? mbe : rbe;
        z[b][j] = dense[b * 16 + j];
        __syncthreads();
        for (int l = 0; l < 3; ++l) {
            float s = Bv[l * 16 + j];
            for (int i = 0; i < 16; ++i) s += z[b][i] * W[l * 256 + i * 16 + j];
            t2[b][j] = s;
            __syncthreads();
            float mu = 0.0f;
            for (int i = 0; i < 16; ++i) mu += t2[b][i];
            mu *= (1.0f / 16.0f);
            float var = 0.0f;
            for (int i = 0; i < 16; ++i) { float d = t2[b][i] - mu; var += d * d; }
            var *= (1.0f / 16.0f);
            float zn = (s - mu) * rsqrtf(var + 1e-5f) * G[l * 16 + j] + BE[l * 16 + j];
            zn = zn > 0.0f ? zn : 0.2f * zn;
            __syncthreads();
            z[b][j] = zn;
            __syncthreads();
        }
        if (which == 0) {
            float o = 0.0f;
            if (j < 8) {
                o = rob[j];
                for (int i = 0; i < 16; ++i) o += z[b][i] * row_[i * 8 + j];
            }
            t2[b][j] = o;
            __syncthreads();
            if (j == 0) {
                float mx = -3.4e38f;
                for (int r = 0; r < 8; ++r) mx = fmaxf(mx, t2[b][r]);
                float se = 0.0f;
                for (int r = 0; r < 8; ++r) se += expf(t2[b][r] - mx);
                for (int r = 0; r < 8; ++r)
                    room[b * 8 + r] = expf(t2[b][r] - mx) / se;
            }
            __syncthreads();
        } else {
            if (j == 0) {
                float s = mob[0];
                for (int i = 0; i < 16; ++i) s += z[b][i] * mow[i];
                mix[b] = 1.0f / (1.0f + expf(-s));
            }
            __syncthreads();
        }
    }
}

// ---------------------------------------------------------------------------
// 8. impulse[b][n] = sum_r room[b][r] * impulses[r][n]
// ---------------------------------------------------------------------------
__global__ void __launch_bounds__(256) impulse_kernel(
    const float* __restrict__ room, const float* __restrict__ impulses,
    float* __restrict__ imp)
{
    const int b = blockIdx.y;
    const int n = blockIdx.x * 256 + threadIdx.x;
    float s = 0.0f;
#pragma unroll
    for (int r = 0; r < 8; ++r)
        s += room[b * 8 + r] * impulses[(size_t)r * NSAMP + n];
    imp[(size_t)b * NSAMP + n] = s;
}

// ---------------------------------------------------------------------------
// 9. wet[n] = sum_{m<=n} dry[m] * imp[n-m] (causal direct conv, LDS-tiled,
//    async-staged dry tile), fused: out = dry*mix + wet*(1-mix)
// ---------------------------------------------------------------------------
__global__ void __launch_bounds__(256) wet_mix_kernel(
    const float* __restrict__ dry, const float* __restrict__ imp,
    const float* __restrict__ mix, float* __restrict__ out)
{
    __shared__ float ds[256];
    __shared__ float is[512];
    const int b = blockIdx.y;
    const int n0 = blockIdx.x * 256;
    const int tid = threadIdx.x;
    const int n = n0 + tid;
    const float* db = dry + (size_t)b * NSAMP;
    const float* ib = imp + (size_t)b * NSAMP;
    float acc = 0.0f;

    for (int m0 = 0; m0 <= n0; m0 += 256) {
        __syncthreads();
#if USE_ASYNC
        __builtin_amdgcn_global_load_async_to_lds_b32(
            (g_int*)(db + m0 + tid), (l_int*)&ds[tid], 0, 0);
#else
        ds[tid] = db[m0 + tid];
#endif
        const int base = n0 - m0 - 255;
        int i1 = base + tid;
        is[tid] = (i1 >= 0) ? ib[i1] : 0.0f;
        int i2 = i1 + 256;
        is[tid + 256] = (i2 >= 0 && i2 < NSAMP) ? ib[i2] : 0.0f;
#if USE_ASYNC
        __builtin_amdgcn_s_wait_asynccnt(0);
#endif
        __syncthreads();
        int mmax = n - m0; if (mmax > 255) mmax = 255;
        for (int mm = 0; mm <= mmax; ++mm)
            acc = fmaf(ds[mm], is[tid + 255 - mm], acc);
    }
    float mx = mix[b];
    out[(size_t)b * NSAMP + n] = db[n] * mx + acc * (1.0f - mx);
}

// ---------------------------------------------------------------------------
extern "C" void kernel_launch(void* const* d_in, const int* in_sizes, int n_in,
                              void* d_out, int out_size, void* d_ws, size_t ws_size,
                              hipStream_t stream)
{
    const float* x        = (const float*)d_in[0];
    const float* atoms    = (const float*)d_in[1];
    const float* enc_w    = (const float*)d_in[2];
    const float* enc_b    = (const float*)d_in[3];
    const float* up_w     = (const float*)d_in[4];
    const float* up_b     = (const float*)d_in[5];
    const float* lat_w    = (const float*)d_in[6];
    const float* lat_b    = (const float*)d_in[7];
    const float* room_w   = (const float*)d_in[8];
    const float* room_b   = (const float*)d_in[9];
    const float* room_g   = (const float*)d_in[10];
    const float* room_be  = (const float*)d_in[11];
    const float* room_ow  = (const float*)d_in[12];
    const float* room_ob  = (const float*)d_in[13];
    const float* mix_w    = (const float*)d_in[14];
    const float* mix_b    = (const float*)d_in[15];
    const float* mix_g    = (const float*)d_in[16];
    const float* mix_be   = (const float*)d_in[17];
    const float* mix_ow   = (const float*)d_in[18];
    const float* mix_ob   = (const float*)d_in[19];
    const float* impulses = (const float*)d_in[20];
    float* outp = (float*)d_out;

    // workspace layout (floats), ~5.8 MB total
    float* f      = (float*)d_ws;
    float* h0     = f;                       // 4*1024*128
    float* h1     = h0 + 4 * NCH * NFRM;
    float* encout = h1 + 4 * NCH * NFRM;     // 4*256*128
    float* dense  = encout + 4 * 256 * NFRM; // 64
    float* vals   = dense + 64;              // 64
    float* roomp  = vals + 64;               // 32
    float* mixp   = roomp + 32;              // 4
    float* imp    = mixp + 4;                // 4*32768
    float* dryb   = imp + 4 * NSAMP;         // 4*32768
    int*   chs    = (int*)(dryb + 4 * NSAMP);
    int*   ts     = chs + 64;

    stft_kernel<<<dim3(NFRM, 4), 256, 0, stream>>>(x, h0);

    const int dil[8] = {1, 2, 4, 8, 16, 32, 64, 1};
    for (int l = 0; l < 8; ++l) {
        const float* hi = (l & 1) ? h1 : h0;
        float*       ho = (l & 1) ? h0 : h1;
        enc_layer_kernel<<<dim3(8, 4), 256, 0, stream>>>(
            hi, enc_w + (size_t)l * NCH * NCH * 2, enc_b + l * NCH, ho, dil[l]);
    }
    // after 8 layers, current h is h0

    latent_kernel<<<4, 256, 0, stream>>>(h0, lat_w, lat_b, dense);
    proj_wmma_kernel<<<dim3(2, 4), 256, 0, stream>>>(h0, up_w, up_b, encout);
    topk_kernel<<<4, 256, 0, stream>>>(encout, vals, chs, ts);
    dry_kernel<<<dim3(NSAMP / 256, 4), 256, 0, stream>>>(atoms, vals, chs, ts, dryb);
    mlp_kernel<<<1, 64, 0, stream>>>(dense,
        room_w, room_b, room_g, room_be, room_ow, room_ob,
        mix_w, mix_b, mix_g, mix_be, mix_ow, mix_ob, roomp, mixp);
    impulse_kernel<<<dim3(NSAMP / 256, 4), 256, 0, stream>>>(roomp, impulses, imp);
    wet_mix_kernel<<<dim3(NSAMP / 256, 4), 256, 0, stream>>>(dryb, imp, mixp, outp);

    (void)in_sizes; (void)n_in; (void)out_size; (void)ws_size;
}
